// TransformerDecoderLayer_47356309406361
// MI455X (gfx1250) — matrix operand order repaired and checked by reading, not compiled, not run on hardware
//
#include <hip/hip_runtime.h>
#include <hip/hip_bf16.h>

typedef _Float16 f16;
typedef __attribute__((ext_vector_type(16))) _Float16 v16h;
typedef __attribute__((ext_vector_type(8)))  _Float16 v8h;
typedef __attribute__((ext_vector_type(8)))  float    v8f;
typedef __attribute__((ext_vector_type(4)))  unsigned int u32x4;
typedef __attribute__((ext_vector_type(4)))  int i32x4;
typedef __attribute__((ext_vector_type(8)))  int i32x8;

#define CAT16(lo,hi) __builtin_shufflevector(lo,hi,0,1,2,3,4,5,6,7,8,9,10,11,12,13,14,15)

__device__ __forceinline__ v8f wmma_f16f32(v16h a, v16h b, v8f c) {
  // (neg_a, A, neg_b, B, c_mod, C, reuse_a, reuse_b)
  return __builtin_amdgcn_wmma_f32_16x16x32_f16(false, a, false, b, (short)0, c, false, false);
}

__device__ __forceinline__ unsigned sgpr_u32(unsigned v) {
  return (unsigned)__builtin_amdgcn_readfirstlane((int)v);
}

// Issue one 2D TENSOR_LOAD_TO_LDS: tile_d1 rows x tile_d0 f16 elements, row
// stride stride_d0 elements in memory; LDS rows padded 64B data + 16B pad
// (pad_interval=16 DWORDs -> code 3, pad_amount=4 DWORDs -> code 3).
__device__ __forceinline__ void tdm_load_2d(const void* gptr, unsigned lds_byte_off,
                                            unsigned tile_d0, unsigned tile_d1,
                                            unsigned tens_d0, unsigned tens_d1,
                                            unsigned stride_d0) {
  unsigned long long ga = (unsigned long long)gptr;
  u32x4 g0;
  g0[0] = sgpr_u32(1u);                                   // count=1, user mode
  g0[1] = sgpr_u32(lds_byte_off);                         // lds_addr
  g0[2] = sgpr_u32((unsigned)(ga & 0xFFFFFFFFu));         // global_addr[31:0]
  g0[3] = sgpr_u32((unsigned)((ga >> 32) & 0x01FFFFFFu) | (2u << 30)); // addr[56:32] | type=2
  i32x8 g1;
  g1[0] = (int)sgpr_u32((1u << 16) | (1u << 20) | (3u << 22) | (3u << 25)); // 2B elems, pad en
  g1[1] = (int)sgpr_u32((tens_d0 & 0xFFFFu) << 16);                         // tensor_dim0 lo
  g1[2] = (int)sgpr_u32(((tens_d0 >> 16) & 0xFFFFu) | ((tens_d1 & 0xFFFFu) << 16));
  g1[3] = (int)sgpr_u32(((tens_d1 >> 16) & 0xFFFFu) | ((tile_d0 & 0xFFFFu) << 16));
  g1[4] = (int)sgpr_u32(tile_d1 & 0xFFFFu);               // tile_dim1 (tile_dim2=0)
  g1[5] = (int)sgpr_u32(stride_d0);                       // tensor_dim0_stride[31:0]
  g1[6] = (int)sgpr_u32(0u);                              // stride hi / dim1_stride lo
  g1[7] = (int)sgpr_u32(0u);
  const i32x4 z4 = {0,0,0,0};
#if __clang_major__ >= 23
  const i32x8 z8 = {0,0,0,0,0,0,0,0};
  __builtin_amdgcn_tensor_load_to_lds(g0, g1, z4, z4, z8, 0);
#else
  __builtin_amdgcn_tensor_load_to_lds(g0, g1, z4, z4, 0);
#endif
}

// ---------------------------------------------------------------- casts
__global__ void cast_x4_kernel(const float* __restrict__ src, f16* __restrict__ dst, int n4) {
  int i = blockIdx.x * blockDim.x + threadIdx.x;
  if (i < n4) {
    float4 v = ((const float4*)src)[i];
    f16* d = dst + (size_t)i * 4;
    d[0] = (f16)v.x; d[1] = (f16)v.y; d[2] = (f16)v.z; d[3] = (f16)v.w;
  }
}

// W[k][n] (f32) -> WT[n][k] (f16)
__global__ void cast_tr_kernel(const float* __restrict__ W, f16* __restrict__ WT, int K, int N) {
  int i = blockIdx.x * blockDim.x + threadIdx.x;
  if (i < K * N) {
    int k = i / N, n = i % N;
    WT[(size_t)n * K + k] = (f16)W[i];
  }
}

// ---------------------------------------------------------------- WMMA GEMM
// C[M][N] = A[M][K](f16) * BT[N][K](f16)^T + bias ; optional residual / relu.
// 128x128 tile, 256 threads = 8 waves (2x4); wave tile 64x32 -> 4x2 fragments.
// A tile staged with vector loads, B tile staged with the Tensor Data Mover.
template<bool RELU, bool HAS_RES, bool OUT32, bool OUT16>
__global__ void __launch_bounds__(256)
gemm_wmma(const f16* __restrict__ A, const f16* __restrict__ BT,
          const float* __restrict__ bias, const float* __restrict__ res,
          float* __restrict__ out32, f16* __restrict__ out16,
          int M, int N, int K)
{
  __shared__ __align__(16) f16 Al[128][40];   // +8 pad: conflict-free frag reads
  __shared__ __align__(16) f16 Bl[128][40];   // TDM writes 64B rows + 16B pad

  const int tid  = threadIdx.x;
  const int lane = tid & 31;
  const int w    = tid >> 5;
  const int wm   = w >> 2;        // 0..1
  const int wn   = w & 3;         // 0..3
  const int ln   = lane & 15;
  const int hf   = lane >> 4;     // wave32 half
  const int m0   = blockIdx.y * 128;
  const int n0   = blockIdx.x * 128;

  v8f acc[4][2];
  const v8f vzero = {0.f,0.f,0.f,0.f,0.f,0.f,0.f,0.f};
#pragma unroll
  for (int i = 0; i < 4; i++)
#pragma unroll
    for (int j = 0; j < 2; j++) acc[i][j] = vzero;

  // per-wave TDM destination (16 rows of Bl) -- flat LDS addr[31:0] == LDS offset
  const unsigned bl_lds = (unsigned)(unsigned long long)&Bl[w * 16][0];

  for (int k0 = 0; k0 < K; k0 += 32) {
    __syncthreads();
    // stage A tile [128][32]: 16B per thread-chunk, 2 chunks each
#pragma unroll
    for (int c = 0; c < 2; c++) {
      int cid = tid + c * 256;
      int row = cid >> 2;
      int col = (cid & 3) * 8;
      *(v8h*)&Al[row][col] = *(const v8h*)&A[(size_t)(m0 + row) * K + k0 + col];
    }
    // stage B tile: each wave DMAs its 16x32 slice via TDM
    tdm_load_2d(&BT[(size_t)(n0 + w * 16) * K + k0], bl_lds,
                /*tile*/ 32u, 16u,
                /*tensor extent from tile start*/ (unsigned)(K - k0),
                (unsigned)(N - (n0 + w * 16)),
                /*stride*/ (unsigned)K);
    if (k0 + 32 < K) {  // hint next tiles toward L2 (global_prefetch_b8)
      __builtin_prefetch(&A [(size_t)(m0 + (tid >> 1)) * K + k0 + 32], 0, 1);
      __builtin_prefetch(&BT[(size_t)(n0 + (tid >> 1)) * K + k0 + 32], 0, 1);
    }
    __builtin_amdgcn_s_wait_tensorcnt((short)0);   // our slice is in LDS
    __syncthreads();                               // everyone's slices are in LDS

    v16h af[4], bf[2];
#pragma unroll
    for (int fm = 0; fm < 4; fm++) {
      int m = wm * 64 + fm * 16 + ln;                 // A layout: row=lane%16
      v8h lo = *(const v8h*)&Al[m][8 * hf];           // e<8 : K = e + 8*half
      v8h hi = *(const v8h*)&Al[m][16 + 8 * hf];      // e>=8: K = e + 8 + 8*half
      af[fm] = CAT16(lo, hi);
    }
#pragma unroll
    for (int fn = 0; fn < 2; fn++) {
      int n = wn * 32 + fn * 16 + ln;                 // B layout: col=lane%16
      v8h lo = *(const v8h*)&Bl[n][8 * hf];           // e<8 : K = e + 8*half
      v8h hi = *(const v8h*)&Bl[n][16 + 8 * hf];      // e>=8: K = e + 8 + 8*half
      bf[fn] = CAT16(lo, hi);
    }
#pragma unroll
    for (int fm = 0; fm < 4; fm++)
#pragma unroll
      for (int fn = 0; fn < 2; fn++)
        acc[fm][fn] = wmma_f16f32(af[fm], bf[fn], acc[fm][fn]);
  }

  // epilogue: D layout row m = r + 8*half, col n = lane%16
#pragma unroll
  for (int fm = 0; fm < 4; fm++) {
#pragma unroll
    for (int fn = 0; fn < 2; fn++) {
      int n = n0 + wn * 32 + fn * 16 + ln;
      float bv = bias[n];
#pragma unroll
      for (int r = 0; r < 8; r++) {
        int m = m0 + wm * 64 + fm * 16 + r + 8 * hf;
        float v = acc[fm][fn][r] + bv;
        if constexpr (HAS_RES) v += res[(size_t)m * N + n];
        if constexpr (RELU)    v = v > 0.f ? v : 0.f;
        if constexpr (OUT32)   out32[(size_t)m * N + n] = v;
        if constexpr (OUT16)   out16[(size_t)m * N + n] = (f16)v;
      }
    }
  }
}

// ---------------------------------------------------------------- flash attention
// Q/K/V are f16 in [B*S][H*64] layout. Block: 128 queries of one (b,h), 4 waves x 32 q.
// Key tiles of 64; online softmax; WMMA for both QK^T and PV.
__global__ void __launch_bounds__(128)
attn_kernel(const f16* __restrict__ Q, const f16* __restrict__ Kb,
            const f16* __restrict__ Vb, f16* __restrict__ O, int S)
{
  __shared__ __align__(16) f16 Kl[64][72];    // [key][dk]
  __shared__ __align__(16) f16 Vt[64][72];    // [dk][key]  (transposed at staging)
  __shared__ __align__(16) f16 Pl[128][72];   // probs, wave-private 32-row bands

  const int tid  = threadIdx.x;
  const int lane = tid & 31;
  const int w    = tid >> 5;      // 0..3
  const int ln   = lane & 15;
  const int hf   = lane >> 4;
  const int qtile = blockIdx.x;
  const int bh    = blockIdx.y;   // b*H + h
  const int bb    = bh >> 4;
  const int h     = bh & 15;
  const size_t ld = 1024;                           // H*DK
  const size_t base = (size_t)bb * S * ld + (size_t)h * 64;
  const int qb = qtile * 128;
  const int qw = qb + w * 32;

  // Q fragments (A operand), kept live: [row-frag][k-step]
  v16h qf[2][2];
#pragma unroll
  for (int fm = 0; fm < 2; fm++) {
    int q = qw + fm * 16 + ln;
    const f16* qrow = Q + base + (size_t)q * ld;
#pragma unroll
    for (int kk = 0; kk < 2; kk++) {
      v8h lo = *(const v8h*)(qrow + kk * 32 + 8 * hf);
      v8h hi = *(const v8h*)(qrow + kk * 32 + 16 + 8 * hf);
      qf[fm][kk] = CAT16(lo, hi);
    }
  }

  const v8f vzero = {0.f,0.f,0.f,0.f,0.f,0.f,0.f,0.f};
  v8f o[2][4];
#pragma unroll
  for (int i = 0; i < 2; i++)
#pragma unroll
    for (int j = 0; j < 4; j++) o[i][j] = vzero;

  float mst[2][8], lst[2][8];
#pragma unroll
  for (int i = 0; i < 2; i++)
#pragma unroll
    for (int r = 0; r < 8; r++) { mst[i][r] = -1e30f; lst[i][r] = 0.f; }

  const int njt = qtile * 2 + 2;                    // causal tile bound
  for (int j = 0; j < njt; j++) {
    const int kb = j * 64;
    __syncthreads();
    // stage K (as-is) and V (transposed): 64 rows x 64 f16, 4 chunks/thread
#pragma unroll
    for (int c = 0; c < 4; c++) {
      int cid = tid + c * 128;
      int row = cid >> 3;           // 0..63
      int col = (cid & 7) * 8;      // 0..56
      *(v8h*)&Kl[row][col] = *(const v8h*)&Kb[base + (size_t)(kb + row) * ld + col];
      v8h vv = *(const v8h*)&Vb[base + (size_t)(kb + row) * ld + col];
#pragma unroll
      for (int e = 0; e < 8; e++) Vt[col + e][row] = vv[e];
    }
    __syncthreads();

    const bool need_mask = (j >= 2 * qtile);

    // preload all K-tile B fragments so the 16 QK^T WMMAs can burst
    v16h kf[4][2];
#pragma unroll
    for (int fn = 0; fn < 4; fn++) {
#pragma unroll
      for (int kk = 0; kk < 2; kk++) {
        int n = fn * 16 + ln;                               // key (local)
        v8h lo = *(const v8h*)&Kl[n][kk * 32 + 8 * hf];     // table-order dk
        v8h hi = *(const v8h*)&Kl[n][kk * 32 + 16 + 8 * hf];
        kf[fn][kk] = CAT16(lo, hi);
      }
    }

#pragma unroll
    for (int fm = 0; fm < 2; fm++) {
      // S = Q K^T  (contraction over dk=64, two k-steps)
      v8f s[4];
#pragma unroll
      for (int fn = 0; fn < 4; fn++) {
        v8f z = vzero;
#pragma unroll
        for (int kk = 0; kk < 2; kk++)
          z = wmma_f16f32(qf[fm][kk], kf[fn][kk], z);
        s[fn] = z;
      }
      // scale + causal mask
#pragma unroll
      for (int fn = 0; fn < 4; fn++) {
        int key = kb + fn * 16 + ln;
#pragma unroll
        for (int r = 0; r < 8; r++) {
          float v = s[fn][r] * 0.125f;                      // 1/sqrt(64)
          if (need_mask) {
            int q = qw + fm * 16 + r + 8 * hf;
            if (key > q) v = -1e30f;
          }
          s[fn][r] = v;
        }
      }
      // online softmax per row; rows of a half live in lanes [hf*16, hf*16+16)
#pragma unroll
      for (int r = 0; r < 8; r++) {
        float rmax = s[0][r];
        rmax = fmaxf(rmax, s[1][r]);
        rmax = fmaxf(rmax, s[2][r]);
        rmax = fmaxf(rmax, s[3][r]);
#pragma unroll
        for (int off = 1; off < 16; off <<= 1)
          rmax = fmaxf(rmax, __shfl_xor(rmax, off, 32));
        float mo = mst[fm][r];
        float mn = fmaxf(mo, rmax);
        float alpha = __expf(mo - mn);
        float rsum = 0.f;
#pragma unroll
        for (int fn = 0; fn < 4; fn++) {
          float p = __expf(s[fn][r] - mn);
          s[fn][r] = p;
          rsum += p;
        }
#pragma unroll
        for (int off = 1; off < 16; off <<= 1)
          rsum += __shfl_xor(rsum, off, 32);
        lst[fm][r] = lst[fm][r] * alpha + rsum;
        mst[fm][r] = mn;
#pragma unroll
        for (int fo = 0; fo < 4; fo++) o[fm][fo][r] *= alpha;
        // spill P (D layout -> LDS), wave-private rows: no barrier needed
        int prow = w * 32 + fm * 16 + r + 8 * hf;
#pragma unroll
        for (int fn = 0; fn < 4; fn++) Pl[prow][fn * 16 + ln] = (f16)s[fn][r];
      }
    }

    // preload all V fragments (kf is dead now), then burst the 16 PV WMMAs
    v16h vf[4][2];
#pragma unroll
    for (int fo = 0; fo < 4; fo++) {
#pragma unroll
      for (int kk = 0; kk < 2; kk++) {
        int n = fo * 16 + ln;                               // dk (local)
        v8h lo = *(const v8h*)&Vt[n][kk * 32 + 8 * hf];     // table-order key
        v8h hi = *(const v8h*)&Vt[n][kk * 32 + 16 + 8 * hf];
        vf[fo][kk] = CAT16(lo, hi);
      }
    }
#pragma unroll
    for (int fm = 0; fm < 2; fm++) {
      v16h pf[2];
#pragma unroll
      for (int kk = 0; kk < 2; kk++) {
        int m = w * 32 + fm * 16 + ln;
        v8h lo = *(const v8h*)&Pl[m][kk * 32 + 8 * hf];
        v8h hi = *(const v8h*)&Pl[m][kk * 32 + 16 + 8 * hf];
        pf[kk] = CAT16(lo, hi);
      }
#pragma unroll
      for (int fo = 0; fo < 4; fo++)
#pragma unroll
        for (int kk = 0; kk < 2; kk++)
          o[fm][fo] = wmma_f16f32(pf[kk], vf[fo][kk], o[fm][fo]);
    }
  }

  // normalize + store f16 in [B*S][H*64]
#pragma unroll
  for (int fm = 0; fm < 2; fm++) {
#pragma unroll
    for (int r = 0; r < 8; r++) {
      float inv = 1.0f / lst[fm][r];
      int q = qw + fm * 16 + r + 8 * hf;
      f16* orow = O + base + (size_t)q * ld;
#pragma unroll
      for (int fo = 0; fo < 4; fo++)
        orow[fo * 16 + ln] = (f16)(o[fm][fo][r] * inv);
    }
  }
}

// ---------------------------------------------------------------- launch
extern "C" void kernel_launch(void* const* d_in, const int* in_sizes, int n_in,
                              void* d_out, int out_size, void* d_ws, size_t ws_size,
                              hipStream_t stream) {
  (void)in_sizes; (void)n_in; (void)out_size; (void)ws_size;
  const float* x  = (const float*)d_in[0];
  const float* Wq = (const float*)d_in[1];
  const float* bq = (const float*)d_in[2];
  const float* Wk = (const float*)d_in[3];
  const float* bk = (const float*)d_in[4];
  const float* Wv = (const float*)d_in[5];
  const float* bv = (const float*)d_in[6];
  const float* Wo = (const float*)d_in[7];
  const float* bo = (const float*)d_in[8];
  const float* W1 = (const float*)d_in[9];
  const float* b1 = (const float*)d_in[10];
  const float* W2 = (const float*)d_in[11];
  const float* b2 = (const float*)d_in[12];

  const int S = 2048, D = 1024, FF = 4096;
  const int M = 2 * S;                   // B*S = 4096
  const size_t MB = 1u << 20;
  char* ws = (char*)d_ws;
  f16*   x16   = (f16*)(ws + 0);         // 8 MB; later reused as attn16, then h16
  f16*   q16   = (f16*)(ws + 8  * MB);
  f16*   k16   = (f16*)(ws + 16 * MB);
  f16*   v16   = (f16*)(ws + 24 * MB);
  f16*   attn16= x16;                    // alias (x16 dead after QKV GEMMs)
  f16*   h16   = (f16*)(ws + 0);         // 32 MB, alias (q/k/v/attn dead by FFN1)
  f16*   WqT   = (f16*)(ws + 32 * MB);
  f16*   WkT   = (f16*)(ws + 34 * MB);
  f16*   WvT   = (f16*)(ws + 36 * MB);
  f16*   WoT   = (f16*)(ws + 38 * MB);
  f16*   W1T   = (f16*)(ws + 40 * MB);   // 8 MB
  f16*   W2T   = (f16*)(ws + 48 * MB);   // 8 MB
  float* x1    = (float*)(ws + 56 * MB); // 16 MB f32 (x + attn_out)
  f16*   x1h   = (f16*)(ws + 72 * MB);   // 8 MB

  // casts
  cast_x4_kernel<<<(M * D / 4 + 255) / 256, 256, 0, stream>>>(x, x16, M * D / 4);
  cast_tr_kernel<<<(D * D + 255) / 256, 256, 0, stream>>>(Wq, WqT, D, D);
  cast_tr_kernel<<<(D * D + 255) / 256, 256, 0, stream>>>(Wk, WkT, D, D);
  cast_tr_kernel<<<(D * D + 255) / 256, 256, 0, stream>>>(Wv, WvT, D, D);
  cast_tr_kernel<<<(D * D + 255) / 256, 256, 0, stream>>>(Wo, WoT, D, D);
  cast_tr_kernel<<<(D * FF + 255) / 256, 256, 0, stream>>>(W1, W1T, D, FF);
  cast_tr_kernel<<<(FF * D + 255) / 256, 256, 0, stream>>>(W2, W2T, FF, D);

  // QKV projections (f16 out)
  dim3 gQKV(D / 128, M / 128);
  gemm_wmma<false,false,false,true><<<gQKV, 256, 0, stream>>>(x16, WqT, bq, nullptr, nullptr, q16, M, D, D);
  gemm_wmma<false,false,false,true><<<gQKV, 256, 0, stream>>>(x16, WkT, bk, nullptr, nullptr, k16, M, D, D);
  gemm_wmma<false,false,false,true><<<gQKV, 256, 0, stream>>>(x16, WvT, bv, nullptr, nullptr, v16, M, D, D);

  // causal flash attention -> attn16 [B*S][1024]
  attn_kernel<<<dim3(S / 128, 2 * 16), 128, 0, stream>>>(q16, k16, v16, attn16, S);

  // x1 = x + attn @ Wo + bo  (f32 + f16 copies)
  gemm_wmma<false,true,true,true><<<gQKV, 256, 0, stream>>>(attn16, WoT, bo, x, x1, x1h, M, D, D);

  // h = relu(x1 @ W1 + b1)
  gemm_wmma<true,false,false,true><<<dim3(FF / 128, M / 128), 256, 0, stream>>>(x1h, W1T, b1, nullptr, nullptr, h16, M, FF, D);

  // out = x1 + h @ W2 + b2
  gemm_wmma<false,true,true,false><<<gQKV, 256, 0, stream>>>(h16, W2T, b2, x1, (float*)d_out, nullptr, M, D, FF);
}
